// Encoder_9148280340798
// MI455X (gfx1250) — matrix operand order repaired and checked by reading, not compiled, not run on hardware
//
#include <hip/hip_runtime.h>

// ---------------------------------------------------------------------------
// flan-t5-large encoder forward on gfx1250 (CDNA5)
//   - bf16 WMMA (v_wmma_f32_16x16x32_bf16), f32 accumulate
//   - double-buffered LDS GEMM pipeline
//   - A-tiles staged with GLOBAL_LOAD_ASYNC_TO_LDS_B128 (ASYNCcnt-tracked)
// ---------------------------------------------------------------------------

typedef __bf16 bf16_t;
typedef __attribute__((ext_vector_type(16))) __bf16 v16bf;
typedef __attribute__((ext_vector_type(8)))  __bf16 v8bf;
typedef __attribute__((ext_vector_type(8)))  float  v8f;

union V16 { v16bf v; v8bf h[2]; };

#define AS3 __attribute__((address_space(3)))

#define NB_L 24
#define NB_D 1024
#define NB_H 16
#define NB_DH 64
#define NB_F 2816
#define NB_B 2
#define NB_S 512
#define NB_M (NB_B * NB_S)        // 1024 tokens
#define HD   (NB_H * NB_DH)       // 1024

// ---------------------------------------------------------------------------
// Generic tiled GEMM: C[M,N](f32) = A[M,K](bf16) x B[K,N](f32->bf16 or bf16)
// Tile 128x64x32; 8 waves (4x2), each wave 2x2 WMMA 16x16 tiles.
// Double-buffered LDS; A staged via async-to-LDS copies.
// ---------------------------------------------------------------------------
template<bool B_F32, bool ADD_RES>
__global__ __launch_bounds__(256) void gemm_bf16_kernel(
    const bf16_t* A, const void* Bv, float* C, const float* Res,
    int M, int N, int K, int lda, int ldb, int ldc,
    long sA, long sB, long sC)
{
    const long z = blockIdx.z;
    A += z * sA;
    C += z * sC;
    const float*  Rp = ADD_RES ? (Res + z * sC) : nullptr;
    const float*  Bf = nullptr;
    const bf16_t* Bh = nullptr;
    if constexpr (B_F32) Bf = (const float*)Bv + z * sB;
    else                 Bh = (const bf16_t*)Bv + z * sB;

    __shared__ bf16_t As[2][128 * 32];   // As[buf][m][k]
    __shared__ bf16_t Bs[2][64 * 32];    // Bs[buf][n][k] (transposed)

    const int tid   = threadIdx.x;
    const int lane  = tid & 31;
    const int wave  = tid >> 5;
    const int wm    = wave & 3;       // wave row (M)
    const int wn    = wave >> 2;      // wave col (N)
    const int lrow  = lane & 15;
    const int lhalf = lane >> 4;

    const int m0 = blockIdx.y * 128;
    const int n0 = blockIdx.x * 64;

    v8f acc[2][2];
    #pragma unroll
    for (int i = 0; i < 2; ++i)
        #pragma unroll
        for (int j = 0; j < 2; ++j)
            #pragma unroll
            for (int e = 0; e < 8; ++e)
                acc[i][j][e] = 0.0f;

    // staging roles
    const int arow = tid >> 1;              // 0..127
    const int acol = (tid & 1) * 16;        // 0 or 16  (elements)
    const int brow = tid >> 3;              // 0..31 (k)
    const int bcol = (tid & 7) * 8;         // 0..56 (n)

    // ---- A tile: 32B/thread via async copy to LDS (offset applies to both) --
    auto asyncA = [&](int buf, int k0) {
        const bf16_t* src = A + (long)(m0 + arow) * lda + k0 + acol;
        unsigned dst = (unsigned)(unsigned long long)
                       (AS3 char*)(char*)&As[buf][arow * 32 + acol];
        asm volatile(
            "global_load_async_to_lds_b128 %0, %1, off\n\t"
            "global_load_async_to_lds_b128 %0, %1, off offset:16"
            :: "v"(dst), "v"(src) : "memory");
    };

    // ---- B tile: load (early) / convert+transpose-store (late) ----
    auto loadB_f32 = [&](int k0, float4& f0, float4& f1) {
        const float* src = Bf + (long)(k0 + brow) * ldb + n0 + bcol;
        f0 = *(const float4*)(src);
        f1 = *(const float4*)(src + 4);
    };
    auto storeB_f32 = [&](int buf, float4 f0, float4 f1) {
        float tmp[8] = {f0.x, f0.y, f0.z, f0.w, f1.x, f1.y, f1.z, f1.w};
        #pragma unroll
        for (int j = 0; j < 8; ++j)
            Bs[buf][(bcol + j) * 32 + brow] = (bf16_t)tmp[j];
    };
    auto loadB_h = [&](int k0, int4& r) {
        r = *(const int4*)(Bh + (long)(k0 + brow) * ldb + n0 + bcol);
    };
    auto storeB_h = [&](int buf, int4 r) {
        union { int4 i; bf16_t e[8]; } u;
        u.i = r;
        #pragma unroll
        for (int j = 0; j < 8; ++j)
            Bs[buf][(bcol + j) * 32 + brow] = u.e[j];
    };

    // ---- fragments + 4 WMMA per 32-K step ----
    auto compute = [&](int buf) {
        // A 16x32: lane m=lane&15; e<8 -> k=8*half+e ; e>=8 -> k=16+8*half+(e-8)
        V16 af[2];
        #pragma unroll
        for (int tm = 0; tm < 2; ++tm) {
            int row = wm * 32 + tm * 16 + lrow;
            af[tm].h[0] = *(const v8bf*)&As[buf][row * 32 + 8 * lhalf];
            af[tm].h[1] = *(const v8bf*)&As[buf][row * 32 + 16 + 8 * lhalf];
        }
        // B 32x16: lane n=lane&15; e<8 -> k=8*half+e ; e>=8 -> k=16+8*half+(e-8)
        V16 bfr[2];
        #pragma unroll
        for (int tn = 0; tn < 2; ++tn) {
            int col = wn * 32 + tn * 16 + lrow;
            bfr[tn].h[0] = *(const v8bf*)&Bs[buf][col * 32 + 8 * lhalf];
            bfr[tn].h[1] = *(const v8bf*)&Bs[buf][col * 32 + 16 + 8 * lhalf];
        }
        #pragma unroll
        for (int tm = 0; tm < 2; ++tm)
            #pragma unroll
            for (int tn = 0; tn < 2; ++tn)
                acc[tm][tn] = __builtin_amdgcn_wmma_f32_16x16x32_bf16(
                    false, af[tm].v, false, bfr[tn].v,
                    (short)0, acc[tm][tn], false, false);
    };

    // ---- prologue: fill buffer 0 ----
    asyncA(0, 0);
    if constexpr (B_F32) {
        float4 f0, f1;
        loadB_f32(0, f0, f1);
        storeB_f32(0, f0, f1);
    } else {
        int4 r;
        loadB_h(0, r);
        storeB_h(0, r);
    }
    asm volatile("s_wait_asynccnt 0" ::: "memory");
    __syncthreads();

    // ---- pipelined main loop ----
    int cur = 0;
    for (int k0 = 0; k0 < K; k0 += 32) {
        const int  nxt  = cur ^ 1;
        const bool more = (k0 + 32) < K;
        float4 f0{}, f1{};
        int4   hreg{};
        if (more) {
            asyncA(nxt, k0 + 32);
            if constexpr (B_F32) {
                loadB_f32(k0 + 32, f0, f1);
                if (k0 + 64 < K)
                    __builtin_prefetch(Bf + (long)(k0 + 64 + brow) * ldb + n0 + bcol, 0, 1);
            } else {
                loadB_h(k0 + 32, hreg);
            }
        }
        compute(cur);                 // WMMAs overlap async copies + B loads
        if (more) {
            if constexpr (B_F32) storeB_f32(nxt, f0, f1);
            else                 storeB_h(nxt, hreg);
        }
        asm volatile("s_wait_asynccnt 0" ::: "memory");
        __syncthreads();
        cur = nxt;
    }

    // ---- epilogue: C/D layout: VGPR r -> m = r + 8*half ; n = lane&15 ----
    #pragma unroll
    for (int tm = 0; tm < 2; ++tm)
        #pragma unroll
        for (int tn = 0; tn < 2; ++tn) {
            int gn = n0 + wn * 32 + tn * 16 + lrow;
            #pragma unroll
            for (int r = 0; r < 8; ++r) {
                int gm = m0 + wm * 32 + tm * 16 + r + 8 * lhalf;
                float vout = acc[tm][tn][r];
                if constexpr (ADD_RES) vout += Rp[(long)gm * ldc + gn];
                C[(long)gm * ldc + gn] = vout;
            }
        }
}

// ---------------------------------------------------------------------------
// Glue kernels
// ---------------------------------------------------------------------------

__global__ void embed_kernel(const int* __restrict__ ids,
                             const float* __restrict__ emb,
                             float* __restrict__ h)
{
    int i = blockIdx.x * blockDim.x + threadIdx.x;   // [M, D] flat, 1<<20
    int row = i >> 10;
    int col = i & 1023;
    h[i] = emb[(long)ids[row] * NB_D + col];
}

// T5 relative-position bias: bias[H, S, S]
__global__ void bias_kernel(const float* __restrict__ rel_table,
                            float* __restrict__ bias)
{
    int i = blockIdx.x * blockDim.x + threadIdx.x;   // S*S
    if (i >= NB_S * NB_S) return;
    int q = i / NB_S;
    int k = i % NB_S;
    int rp = k - q;
    int ret = (rp > 0) ? 16 : 0;
    int arp = rp < 0 ? -rp : rp;
    int bucket;
    if (arp < 8) {
        bucket = ret + arp;
    } else {
        int large = 8 + (int)(logf((float)arp * 0.125f) * (8.0f / logf(16.0f)));
        if (large > 15) large = 15;
        bucket = ret + large;
    }
    #pragma unroll
    for (int hh = 0; hh < NB_H; ++hh)
        bias[(long)hh * NB_S * NB_S + i] = rel_table[bucket * NB_H + hh];
}

// RMSNorm over D=1024; emit bf16 (for GEMM A) or f32 (final output)
template<bool OUT_BF16>
__global__ __launch_bounds__(256) void rmsnorm_kernel(
    const float* __restrict__ x, const float* __restrict__ w, void* out)
{
    int row = blockIdx.x;
    const float* xr = x + (long)row * NB_D;
    float s = 0.0f;
    #pragma unroll
    for (int j = 0; j < 4; ++j) {
        float v = xr[threadIdx.x + j * 256];
        s += v * v;
    }
    __shared__ float red[256];
    red[threadIdx.x] = s;
    __syncthreads();
    for (int off = 128; off > 0; off >>= 1) {
        if (threadIdx.x < off) red[threadIdx.x] += red[threadIdx.x + off];
        __syncthreads();
    }
    float scale = rsqrtf(red[0] * (1.0f / NB_D) + 1e-6f);
    #pragma unroll
    for (int j = 0; j < 4; ++j) {
        int c = threadIdx.x + j * 256;
        float v = xr[c] * scale * w[c];
        if constexpr (OUT_BF16)
            ((bf16_t*)out)[(long)row * NB_D + c] = (bf16_t)v;
        else
            ((float*)out)[(long)row * NB_D + c] = v;
    }
}

// q,k,v f32 [B,S,H,DH] -> qb/vb bf16 [B,H,S,DH], kT bf16 [B,H,DH,S]
__global__ void qkv_reshape_kernel(const float* __restrict__ qf,
                                   const float* __restrict__ kf,
                                   const float* __restrict__ vf,
                                   bf16_t* __restrict__ qb,
                                   bf16_t* __restrict__ kT,
                                   bf16_t* __restrict__ vb)
{
    int i = blockIdx.x * blockDim.x + threadIdx.x;   // 1<<20
    int d = i & 63;
    int h = (i >> 6) & 15;
    int s = (i >> 10) & 511;
    int b = i >> 19;
    long bh = (long)b * NB_H + h;
    qb[(bh * NB_S + s) * NB_DH + d] = (bf16_t)qf[i];
    vb[(bh * NB_S + s) * NB_DH + d] = (bf16_t)vf[i];
    kT[(bh * NB_DH + d) * NB_S + s] = (bf16_t)kf[i];
}

// softmax(scores + bias) -> probs (bf16). One row [512] per 128-thread block.
__global__ __launch_bounds__(128) void softmax_bias_kernel(
    const float* __restrict__ scores, const float* __restrict__ bias,
    bf16_t* __restrict__ probs)
{
    int row = blockIdx.x;                 // [b][h][q], B*H*S rows
    int q = row & 511;
    int h = (row >> 9) & 15;
    const float* sr = scores + (long)row * NB_S;
    const float* br = bias + ((long)h * NB_S + q) * NB_S;
    int tid = threadIdx.x;

    float v[4];
    float mx = -3.0e38f;
    #pragma unroll
    for (int j = 0; j < 4; ++j) {
        int c = tid + j * 128;
        v[j] = sr[c] + br[c];
        mx = fmaxf(mx, v[j]);
    }
    __shared__ float red[128];
    red[tid] = mx;
    __syncthreads();
    for (int off = 64; off > 0; off >>= 1) {
        if (tid < off) red[tid] = fmaxf(red[tid], red[tid + off]);
        __syncthreads();
    }
    mx = red[0];
    __syncthreads();

    float sum = 0.0f;
    #pragma unroll
    for (int j = 0; j < 4; ++j) {
        v[j] = __expf(v[j] - mx);
        sum += v[j];
    }
    red[tid] = sum;
    __syncthreads();
    for (int off = 64; off > 0; off >>= 1) {
        if (tid < off) red[tid] += red[tid + off];
        __syncthreads();
    }
    float inv = 1.0f / red[0];
    #pragma unroll
    for (int j = 0; j < 4; ++j)
        probs[(long)row * NB_S + tid + j * 128] = (bf16_t)(v[j] * inv);
}

// attn f32 [B,H,S,DH] -> attnb bf16 [B,S,H*DH]
__global__ void attn_reshape_kernel(const float* __restrict__ attnf,
                                    bf16_t* __restrict__ attnb)
{
    int i = blockIdx.x * blockDim.x + threadIdx.x;   // dst [b][s][h][d]
    int d = i & 63;
    int h = (i >> 6) & 15;
    int s = (i >> 10) & 511;
    int b = i >> 19;
    attnb[i] = (bf16_t)attnf[(((long)(b * NB_H + h) * NB_S) + s) * NB_DH + d];
}

// gated-gelu (tanh approx): out = gelu(g0) * g1, emitted as bf16
__global__ void gelu_gate_kernel(const float* __restrict__ g0,
                                 const float* __restrict__ g1,
                                 bf16_t* __restrict__ out, int n)
{
    int i = blockIdx.x * blockDim.x + threadIdx.x;
    if (i >= n) return;
    float x = g0[i];
    float g = 0.5f * x * (1.0f + tanhf(0.7978845608028654f * (x + 0.044715f * x * x * x)));
    out[i] = (bf16_t)(g * g1[i]);
}

// ---------------------------------------------------------------------------
// Host orchestration
// ---------------------------------------------------------------------------
extern "C" void kernel_launch(void* const* d_in, const int* in_sizes, int n_in,
                              void* d_out, int out_size, void* d_ws, size_t ws_size,
                              hipStream_t stream)
{
    const int*   ids = (const int*)d_in[0];
    const float* emb = (const float*)d_in[1];
    const float* rel = (const float*)d_in[2];
    const float* ln1 = (const float*)d_in[3];
    const float* qw  = (const float*)d_in[4];
    const float* kw  = (const float*)d_in[5];
    const float* vw  = (const float*)d_in[6];
    const float* ow  = (const float*)d_in[7];
    const float* ln2 = (const float*)d_in[8];
    const float* wi0 = (const float*)d_in[9];
    const float* wi1 = (const float*)d_in[10];
    const float* wo  = (const float*)d_in[11];
    const float* fln = (const float*)d_in[12];

    // workspace carve-up
    char* p = (char*)d_ws;
    auto carve = [&](size_t bytes) {
        void* r = (void*)p;
        p += (bytes + 255) & ~(size_t)255;
        return r;
    };
    float*  h      = (float*)carve((size_t)NB_M * NB_D * 4);
    bf16_t* xb     = (bf16_t*)carve((size_t)NB_M * NB_D * 2);
    float*  qf     = (float*)carve((size_t)NB_M * HD * 4);
    float*  kf     = (float*)carve((size_t)NB_M * HD * 4);
    float*  vf     = (float*)carve((size_t)NB_M * HD * 4);
    bf16_t* qb     = (bf16_t*)carve((size_t)NB_M * HD * 2);
    bf16_t* kT     = (bf16_t*)carve((size_t)NB_M * HD * 2);
    bf16_t* vb     = (bf16_t*)carve((size_t)NB_M * HD * 2);
    float*  scores = (float*)carve((size_t)NB_B * NB_H * NB_S * NB_S * 4);
    bf16_t* probs  = (bf16_t*)carve((size_t)NB_B * NB_H * NB_S * NB_S * 2);
    float*  attnf  = (float*)carve((size_t)NB_M * HD * 4);
    bf16_t* attnb  = (bf16_t*)carve((size_t)NB_M * HD * 2);
    float*  g0     = (float*)carve((size_t)NB_M * NB_F * 4);
    float*  g1     = (float*)carve((size_t)NB_M * NB_F * 4);
    bf16_t* ffb    = (bf16_t*)carve((size_t)NB_M * NB_F * 2);
    float*  biasb  = (float*)carve((size_t)NB_H * NB_S * NB_S * 4);

    (void)in_sizes; (void)n_in; (void)out_size; (void)ws_size;

    // embedding + shared relative-position bias
    embed_kernel<<<dim3(4096), dim3(256), 0, stream>>>(ids, emb, h);
    bias_kernel<<<dim3((NB_S * NB_S + 255) / 256), dim3(256), 0, stream>>>(rel, biasb);

    const dim3 gProj(HD / 64, NB_M / 128, 1);                 // 16 x 8
    const dim3 gScores(NB_S / 64, NB_S / 128, NB_B * NB_H);   // 8 x 4 x 32
    const dim3 gAttn(NB_DH / 64, NB_S / 128, NB_B * NB_H);    // 1 x 4 x 32
    const dim3 gFF(NB_F / 64, NB_M / 128, 1);                 // 44 x 8
    const dim3 gOut(NB_D / 64, NB_M / 128, 1);                // 16 x 8

    for (int l = 0; l < NB_L; ++l) {
        const float* qw_l  = qw  + (long)l * NB_D * HD;
        const float* kw_l  = kw  + (long)l * NB_D * HD;
        const float* vw_l  = vw  + (long)l * NB_D * HD;
        const float* ow_l  = ow  + (long)l * HD * NB_D;
        const float* wi0_l = wi0 + (long)l * NB_D * NB_F;
        const float* wi1_l = wi1 + (long)l * NB_D * NB_F;
        const float* wo_l  = wo  + (long)l * NB_F * NB_D;

        // --- attention ---
        rmsnorm_kernel<true><<<dim3(NB_M), dim3(256), 0, stream>>>(h, ln1 + (long)l * NB_D, xb);

        gemm_bf16_kernel<true, false><<<gProj, 256, 0, stream>>>(
            xb, qw_l, qf, nullptr, NB_M, HD, NB_D, NB_D, HD, HD, 0, 0, 0);
        gemm_bf16_kernel<true, false><<<gProj, 256, 0, stream>>>(
            xb, kw_l, kf, nullptr, NB_M, HD, NB_D, NB_D, HD, HD, 0, 0, 0);
        gemm_bf16_kernel<true, false><<<gProj, 256, 0, stream>>>(
            xb, vw_l, vf, nullptr, NB_M, HD, NB_D, NB_D, HD, HD, 0, 0, 0);

        qkv_reshape_kernel<<<dim3(4096), dim3(256), 0, stream>>>(qf, kf, vf, qb, kT, vb);

        // scores[b,h] = q[S,DH] @ kT[DH,S]
        gemm_bf16_kernel<false, false><<<gScores, 256, 0, stream>>>(
            qb, kT, scores, nullptr,
            NB_S, NB_S, NB_DH, NB_DH, NB_S, NB_S,
            (long)NB_S * NB_DH, (long)NB_DH * NB_S, (long)NB_S * NB_S);

        softmax_bias_kernel<<<dim3(NB_B * NB_H * NB_S), dim3(128), 0, stream>>>(scores, biasb, probs);

        // attn[b,h] = probs[S,S] @ v[S,DH]
        gemm_bf16_kernel<false, false><<<gAttn, 256, 0, stream>>>(
            probs, vb, attnf, nullptr,
            NB_S, NB_DH, NB_S, NB_S, NB_DH, NB_DH,
            (long)NB_S * NB_S, (long)NB_S * NB_DH, (long)NB_S * NB_DH);

        attn_reshape_kernel<<<dim3(4096), dim3(256), 0, stream>>>(attnf, attnb);

        // h += attnb @ o_w  (in-place residual)
        gemm_bf16_kernel<true, true><<<gOut, 256, 0, stream>>>(
            attnb, ow_l, h, h, NB_M, NB_D, HD, HD, NB_D, NB_D, 0, 0, 0);

        // --- FFN ---
        rmsnorm_kernel<true><<<dim3(NB_M), dim3(256), 0, stream>>>(h, ln2 + (long)l * NB_D, xb);

        gemm_bf16_kernel<true, false><<<gFF, 256, 0, stream>>>(
            xb, wi0_l, g0, nullptr, NB_M, NB_F, NB_D, NB_D, NB_F, NB_F, 0, 0, 0);
        gemm_bf16_kernel<true, false><<<gFF, 256, 0, stream>>>(
            xb, wi1_l, g1, nullptr, NB_M, NB_F, NB_D, NB_D, NB_F, NB_F, 0, 0, 0);

        gelu_gate_kernel<<<dim3((NB_M * NB_F + 255) / 256), dim3(256), 0, stream>>>(
            g0, g1, ffb, NB_M * NB_F);

        // h += ffb @ wo  (in-place residual)
        gemm_bf16_kernel<true, true><<<gOut, 256, 0, stream>>>(
            ffb, wo_l, h, h, NB_M, NB_D, NB_F, NB_F, NB_D, NB_D, 0, 0, 0);
    }

    // final RMSNorm -> f32 output
    rmsnorm_kernel<false><<<dim3(NB_M), dim3(256), 0, stream>>>(h, fln, (void*)d_out);
}
